// RNN_4715874091371
// MI455X (gfx1250) — matrix operand order repaired and checked by reading, not compiled, not run on hardware
//
#include <hip/hip_runtime.h>

// Problem sizes (match reference)
#define B_SZ 1024
#define T_SZ 64
#define I_SZ 1024
#define H_SZ 1024
#define KCAT (I_SZ + H_SZ)   // 2048: concatenated K of [x | h] GEMM

typedef __bf16 bf16;
typedef __attribute__((ext_vector_type(16))) __bf16 v16bf;
typedef __attribute__((ext_vector_type(8)))  __bf16 v8bf;
typedef __attribute__((ext_vector_type(8)))  float  v8f;

// ---------------------------------------------------------------------------
// Prep: Wcat[j][0:1024]=bf16(W_ih[j]), Wcat[j][1024:2048]=bf16(W_hh[j]);
//       bias[j]=b_ih[j]+b_hh[j]; zero h0; zero loss accumulator.
// Grid covers exactly H_SZ*KCAT = 2M threads.
// ---------------------------------------------------------------------------
__global__ __launch_bounds__(256)
void rnn_prep_kernel(const float* __restrict__ W_ih, const float* __restrict__ b_ih,
                     const float* __restrict__ W_hh, const float* __restrict__ b_hh,
                     bf16* __restrict__ Wcat, bf16* __restrict__ h0,
                     float* __restrict__ bias, float* __restrict__ lacc) {
  int idx = blockIdx.x * blockDim.x + threadIdx.x;   // 0 .. H_SZ*KCAT-1
  int j = idx >> 11;        // row (output unit)
  int c = idx & (KCAT - 1); // concatenated K index
  float w = (c < I_SZ) ? W_ih[j * I_SZ + c] : W_hh[j * H_SZ + (c - I_SZ)];
  Wcat[idx] = (bf16)w;
  if (idx < B_SZ * H_SZ) h0[idx] = (bf16)0.0f;
  if (idx < H_SZ)        bias[idx] = b_ih[idx] + b_hh[idx];
  if (idx == 0)          *lacc = 0.0f;
}

// Helper: build a 16xK=32 bf16 A-fragment from an fp32 row (converted on the fly).
// Layout (contiguous, matches B): lane half holds K = kc+half*16..+16
// (elems 0-15).
__device__ __forceinline__ v16bf load_a_frag_f32(const float* row, int kc, int half) {
  v8f lo = *(const v8f*)(row + kc + half * 16);
  v8f hi = *(const v8f*)(row + kc + half * 16 + 8);
  v16bf a;
  #pragma unroll
  for (int i = 0; i < 8; ++i) { a[i] = (bf16)lo[i]; a[i + 8] = (bf16)hi[i]; }
  return a;
}

__device__ __forceinline__ v16bf load_a_frag_bf16(const bf16* row, int kc, int half) {
  v8bf lo = *(const v8bf*)(row + kc + half * 16);
  v8bf hi = *(const v8bf*)(row + kc + half * 16 + 8);
  return __builtin_shufflevector(lo, hi, 0, 1, 2, 3, 4, 5, 6, 7,
                                         8, 9, 10, 11, 12, 13, 14, 15);
}

// ---------------------------------------------------------------------------
// Fully parallel precompute: Z[r][h] = av_row[r] . W_ih[h]   (r = b*T + t)
// Wave computes a 32x64 tile (8 accumulators; B-fragments reused across the
// two row groups). 2048 row-blocks x 16 col-blocks = 32768 waves.
// ---------------------------------------------------------------------------
__global__ __launch_bounds__(128)
void rnn_xproj_kernel(const float* __restrict__ av, const bf16* __restrict__ Wcat,
                      bf16* __restrict__ Z) {
  const int lane = threadIdx.x & 31;
  const int wave = threadIdx.x >> 5;
  const int tile = blockIdx.x * 4 + wave;    // 0..32767
  const int rb   = tile >> 4;                // 32-row block (0..2047)
  const int cb   = tile & 15;                // 64-col block (0..15)
  const int m    = lane & 15;
  const int half = lane >> 4;

  const float* xrow0 = av + (size_t)(rb * 32 + m) * I_SZ;
  const float* xrow1 = av + (size_t)(rb * 32 + 16 + m) * I_SZ;
  const bf16* w0 = Wcat + (size_t)(cb * 64 +  0 + m) * KCAT + half * 16;
  const bf16* w1 = Wcat + (size_t)(cb * 64 + 16 + m) * KCAT + half * 16;
  const bf16* w2 = Wcat + (size_t)(cb * 64 + 32 + m) * KCAT + half * 16;
  const bf16* w3 = Wcat + (size_t)(cb * 64 + 48 + m) * KCAT + half * 16;

  v8f c00 = {}, c01 = {}, c02 = {}, c03 = {};
  v8f c10 = {}, c11 = {}, c12 = {}, c13 = {};

  for (int kc = 0; kc < I_SZ; kc += 32) {
    v16bf a0 = load_a_frag_f32(xrow0, kc, half);
    v16bf a1 = load_a_frag_f32(xrow1, kc, half);
    v16bf b0 = *(const v16bf*)(w0 + kc);
    v16bf b1 = *(const v16bf*)(w1 + kc);
    v16bf b2 = *(const v16bf*)(w2 + kc);
    v16bf b3 = *(const v16bf*)(w3 + kc);
    c00 = __builtin_amdgcn_wmma_f32_16x16x32_bf16(false, a0, false, b0, (short)0, c00, false, false);
    c10 = __builtin_amdgcn_wmma_f32_16x16x32_bf16(false, a1, false, b0, (short)0, c10, false, false);
    c01 = __builtin_amdgcn_wmma_f32_16x16x32_bf16(false, a0, false, b1, (short)0, c01, false, false);
    c11 = __builtin_amdgcn_wmma_f32_16x16x32_bf16(false, a1, false, b1, (short)0, c11, false, false);
    c02 = __builtin_amdgcn_wmma_f32_16x16x32_bf16(false, a0, false, b2, (short)0, c02, false, false);
    c12 = __builtin_amdgcn_wmma_f32_16x16x32_bf16(false, a1, false, b2, (short)0, c12, false, false);
    c03 = __builtin_amdgcn_wmma_f32_16x16x32_bf16(false, a0, false, b3, (short)0, c03, false, false);
    c13 = __builtin_amdgcn_wmma_f32_16x16x32_bf16(false, a1, false, b3, (short)0, c13, false, false);
  }

  #pragma unroll
  for (int r = 0; r < 8; ++r) {
    const int lr = half * 8 + r;
    size_t row0 = (size_t)(rb * 32 + lr) * H_SZ;
    size_t row1 = (size_t)(rb * 32 + 16 + lr) * H_SZ;
    Z[row0 + cb * 64 +  0 + m] = (bf16)c00[r];
    Z[row0 + cb * 64 + 16 + m] = (bf16)c01[r];
    Z[row0 + cb * 64 + 32 + m] = (bf16)c02[r];
    Z[row0 + cb * 64 + 48 + m] = (bf16)c03[r];
    Z[row1 + cb * 64 +  0 + m] = (bf16)c10[r];
    Z[row1 + cb * 64 + 16 + m] = (bf16)c11[r];
    Z[row1 + cb * 64 + 32 + m] = (bf16)c12[r];
    Z[row1 + cb * 64 + 48 + m] = (bf16)c13[r];
  }
}

// ---------------------------------------------------------------------------
// Recurrence step (Z-path, K=1024): h_out = tanh( Z[t] + h_in @ W_hh^T + bias )
// 16x64 tile per wave, 1024 waves: maximize parallelism on the critical path.
// ---------------------------------------------------------------------------
__global__ __launch_bounds__(128)
void rnn_step_z_kernel(const bf16* __restrict__ Z, int t,
                       const bf16* __restrict__ Wcat,
                       const float* __restrict__ bias,
                       const bf16* __restrict__ h_in,
                       bf16* __restrict__ h_out) {
  const int lane = threadIdx.x & 31;
  const int wave = threadIdx.x >> 5;
  const int tile = blockIdx.x * 4 + wave;   // 0..1023
  const int rb   = tile >> 4;
  const int cb   = tile & 15;
  const int m    = lane & 15;
  const int half = lane >> 4;

  const bf16* hrow = h_in + (size_t)(rb * 16 + m) * H_SZ;
  const bf16* w0 = Wcat + (size_t)(cb * 64 +  0 + m) * KCAT + I_SZ + half * 16;
  const bf16* w1 = Wcat + (size_t)(cb * 64 + 16 + m) * KCAT + I_SZ + half * 16;
  const bf16* w2 = Wcat + (size_t)(cb * 64 + 32 + m) * KCAT + I_SZ + half * 16;
  const bf16* w3 = Wcat + (size_t)(cb * 64 + 48 + m) * KCAT + I_SZ + half * 16;

  v8f c0 = {}, c1 = {}, c2 = {}, c3 = {};

  #pragma unroll 2
  for (int kc = 0; kc < H_SZ; kc += 32) {
    v16bf a = load_a_frag_bf16(hrow, kc, half);
    v16bf b0 = *(const v16bf*)(w0 + kc);
    v16bf b1 = *(const v16bf*)(w1 + kc);
    v16bf b2 = *(const v16bf*)(w2 + kc);
    v16bf b3 = *(const v16bf*)(w3 + kc);
    c0 = __builtin_amdgcn_wmma_f32_16x16x32_bf16(false, a, false, b0, (short)0, c0, false, false);
    c1 = __builtin_amdgcn_wmma_f32_16x16x32_bf16(false, a, false, b1, (short)0, c1, false, false);
    c2 = __builtin_amdgcn_wmma_f32_16x16x32_bf16(false, a, false, b2, (short)0, c2, false, false);
    c3 = __builtin_amdgcn_wmma_f32_16x16x32_bf16(false, a, false, b3, (short)0, c3, false, false);
  }

  #pragma unroll
  for (int r = 0; r < 8; ++r) {
    const int orow = rb * 16 + half * 8 + r;
    const bf16* zr = Z + ((size_t)orow * T_SZ + t) * H_SZ;  // Z row r = b*T + t
    #pragma unroll
    for (int s = 0; s < 4; ++s) {
      int col = cb * 64 + s * 16 + m;
      float acc = (s == 0) ? c0[r] : (s == 1) ? c1[r] : (s == 2) ? c2[r] : c3[r];
      float v = acc + (float)zr[col] + bias[col];
      h_out[(size_t)orow * H_SZ + col] = (bf16)tanhf(v);
    }
  }
}

// ---------------------------------------------------------------------------
// Fused fallback step (K=2048): used when ws is too small for Z.
// ---------------------------------------------------------------------------
__global__ __launch_bounds__(128)
void rnn_step_fused_kernel(const float* __restrict__ av, int t,
                           const bf16* __restrict__ Wcat,
                           const float* __restrict__ bias,
                           const bf16* __restrict__ h_in,
                           bf16* __restrict__ h_out) {
  const int lane = threadIdx.x & 31;
  const int wave = threadIdx.x >> 5;
  const int tile = blockIdx.x * 4 + wave;
  const int rb   = tile >> 4;
  const int cb   = tile & 15;
  const int m    = lane & 15;
  const int half = lane >> 4;

  const int row = rb * 16 + m;
  const float* xrow = av + (size_t)row * (T_SZ * I_SZ) + (size_t)t * I_SZ;
  const bf16*  hrow = h_in + (size_t)row * H_SZ;
  const bf16* w0 = Wcat + (size_t)(cb * 64 +  0 + m) * KCAT + half * 16;
  const bf16* w1 = Wcat + (size_t)(cb * 64 + 16 + m) * KCAT + half * 16;
  const bf16* w2 = Wcat + (size_t)(cb * 64 + 32 + m) * KCAT + half * 16;
  const bf16* w3 = Wcat + (size_t)(cb * 64 + 48 + m) * KCAT + half * 16;

  v8f c0 = {}, c1 = {}, c2 = {}, c3 = {};

  #pragma unroll 2
  for (int kc = 0; kc < I_SZ; kc += 32) {
    v16bf a = load_a_frag_f32(xrow, kc, half);
    v16bf b0 = *(const v16bf*)(w0 + kc);
    v16bf b1 = *(const v16bf*)(w1 + kc);
    v16bf b2 = *(const v16bf*)(w2 + kc);
    v16bf b3 = *(const v16bf*)(w3 + kc);
    c0 = __builtin_amdgcn_wmma_f32_16x16x32_bf16(false, a, false, b0, (short)0, c0, false, false);
    c1 = __builtin_amdgcn_wmma_f32_16x16x32_bf16(false, a, false, b1, (short)0, c1, false, false);
    c2 = __builtin_amdgcn_wmma_f32_16x16x32_bf16(false, a, false, b2, (short)0, c2, false, false);
    c3 = __builtin_amdgcn_wmma_f32_16x16x32_bf16(false, a, false, b3, (short)0, c3, false, false);
  }
  #pragma unroll 2
  for (int kc = 0; kc < H_SZ; kc += 32) {
    v16bf a = load_a_frag_bf16(hrow, kc, half);
    v16bf b0 = *(const v16bf*)(w0 + I_SZ + kc);
    v16bf b1 = *(const v16bf*)(w1 + I_SZ + kc);
    v16bf b2 = *(const v16bf*)(w2 + I_SZ + kc);
    v16bf b3 = *(const v16bf*)(w3 + I_SZ + kc);
    c0 = __builtin_amdgcn_wmma_f32_16x16x32_bf16(false, a, false, b0, (short)0, c0, false, false);
    c1 = __builtin_amdgcn_wmma_f32_16x16x32_bf16(false, a, false, b1, (short)0, c1, false, false);
    c2 = __builtin_amdgcn_wmma_f32_16x16x32_bf16(false, a, false, b2, (short)0, c2, false, false);
    c3 = __builtin_amdgcn_wmma_f32_16x16x32_bf16(false, a, false, b3, (short)0, c3, false, false);
  }

  #pragma unroll
  for (int r = 0; r < 8; ++r) {
    const int orow = rb * 16 + half * 8 + r;
    #pragma unroll
    for (int s = 0; s < 4; ++s) {
      int col = cb * 64 + s * 16 + m;
      float acc = (s == 0) ? c0[r] : (s == 1) ? c1[r] : (s == 2) ? c2[r] : c3[r];
      h_out[(size_t)orow * H_SZ + col] = (bf16)tanhf(acc + bias[col]);
    }
  }
}

// ---------------------------------------------------------------------------
// Head: out[b] = sigmoid(h[b] . W_out + b_out); accumulate BCE terms.
// ---------------------------------------------------------------------------
__global__ __launch_bounds__(128)
void rnn_out_kernel(const bf16* __restrict__ h, const float* __restrict__ W_out,
                    const float* __restrict__ b_out, const float* __restrict__ y,
                    float* __restrict__ d_out, float* __restrict__ lacc) {
  const int gw   = (blockIdx.x * blockDim.x + threadIdx.x) >> 5;  // batch row
  const int lane = threadIdx.x & 31;
  const bf16* hr = h + (size_t)gw * H_SZ;
  float acc = 0.0f;
  #pragma unroll 4
  for (int k = lane; k < H_SZ; k += 32)
    acc += (float)hr[k] * W_out[k];
  #pragma unroll
  for (int off = 16; off > 0; off >>= 1)
    acc += __shfl_down(acc, off, 32);
  if (lane == 0) {
    float p = 1.0f / (1.0f + expf(-(acc + b_out[0])));
    d_out[1 + gw] = p;
    float yb = (y[gw] >= 1e-5f) ? 1.0f : 0.0f;
    float lp  = fmaxf(logf(p), -100.0f);
    float l1p = fmaxf(log1pf(-p), -100.0f);
    atomicAdd(lacc, yb * lp + (1.0f - yb) * l1p);
  }
}

__global__ void rnn_loss_finalize(const float* __restrict__ lacc,
                                  float* __restrict__ d_out) {
  if (threadIdx.x == 0 && blockIdx.x == 0)
    d_out[0] = -(*lacc) / (float)B_SZ;
}

// ---------------------------------------------------------------------------
// kernel_launch
// Input order: ui(0) uv(1) ai(2) av(3) y(4) W_ih(5) b_ih(6) W_hh(7) b_hh(8)
//              W_out(9) b_out(10).  d_out = [loss, out(1024)].
// ws layout: [0,16MB): Wcat(4MB) | h0(2MB) | h1(2MB) | bias | lacc
//            [16MB, 16MB+128MB): Z (bf16, only if ws_size permits)
// ---------------------------------------------------------------------------
extern "C" void kernel_launch(void* const* d_in, const int* in_sizes, int n_in,
                              void* d_out, int out_size, void* d_ws, size_t ws_size,
                              hipStream_t stream) {
  const float* av    = (const float*)d_in[3];
  const float* y     = (const float*)d_in[4];
  const float* W_ih  = (const float*)d_in[5];
  const float* b_ih  = (const float*)d_in[6];
  const float* W_hh  = (const float*)d_in[7];
  const float* b_hh  = (const float*)d_in[8];
  const float* W_out = (const float*)d_in[9];
  const float* b_out = (const float*)d_in[10];

  bf16*  Wcat = (bf16*)d_ws;
  bf16*  h0   = Wcat + (size_t)H_SZ * KCAT;
  bf16*  h1   = h0 + (size_t)B_SZ * H_SZ;
  float* bias = (float*)(h1 + (size_t)B_SZ * H_SZ);
  float* lacc = bias + H_SZ;
  float* out  = (float*)d_out;

  const size_t z_off   = (size_t)16 * 1024 * 1024;
  const size_t z_bytes = (size_t)T_SZ * B_SZ * H_SZ * sizeof(bf16);  // 128 MB
  const bool   use_z   = (ws_size >= z_off + z_bytes);

  rnn_prep_kernel<<<(H_SZ * KCAT) / 256, 256, 0, stream>>>(
      W_ih, b_ih, W_hh, b_hh, Wcat, h0, bias, lacc);

  bf16* hin = h0;
  bf16* hout = h1;
  if (use_z) {
    bf16* Z = (bf16*)((char*)d_ws + z_off);
    // 32768 tiles / 4 waves per block
    rnn_xproj_kernel<<<8192, 128, 0, stream>>>(av, Wcat, Z);
    for (int t = 0; t < T_SZ; ++t) {
      rnn_step_z_kernel<<<256, 128, 0, stream>>>(Z, t, Wcat, bias, hin, hout);
      bf16* tmp = hin; hin = hout; hout = tmp;
    }
  } else {
    for (int t = 0; t < T_SZ; ++t) {
      rnn_step_fused_kernel<<<256, 128, 0, stream>>>(av, t, Wcat, bias, hin, hout);
      bf16* tmp = hin; hin = hout; hout = tmp;
    }
  }

  rnn_out_kernel<<<(B_SZ * 32) / 128, 128, 0, stream>>>(hin, W_out, b_out, y, out, lacc);
  rnn_loss_finalize<<<1, 32, 0, stream>>>(lacc, out);
}
